// GraphConvAttention_39393440039301
// MI455X (gfx1250) — compile-verified, matrix operand order checked by reading, not run on hardware
//
#include <hip/hip_runtime.h>
#include <hip/hip_bf16.h>

typedef __attribute__((ext_vector_type(16))) _Float16 v16h;
typedef __attribute__((ext_vector_type(8)))  _Float16 v8h;
typedef __attribute__((ext_vector_type(8)))  float    v8f;

#define NEG_SLOPE 0.2f
#define OUT_SLOPE 0.01f

// ---------------------------------------------------------------------------
// Generic zero-fill (32-bit words)
// ---------------------------------------------------------------------------
__global__ void fill_zero_u32(unsigned* __restrict__ p, size_t n) {
    size_t i = (size_t)blockIdx.x * blockDim.x + threadIdx.x;
    size_t stride = (size_t)gridDim.x * blockDim.x;
    for (; i < n; i += stride) p[i] = 0u;
}

// ---------------------------------------------------------------------------
// WMMA GEMM: xl = X @ Wl + bl, xr = X @ Wr + br   (HC <= 16 output columns)
// X rows must have >= (KHALF ? 16 : KPAD) readable floats (zero-padded beyond
// fin). Output row stride 16; cols >= HC written as 0.
// KHALF=true: K in [16,32) of the single 32-block is compile-time zero.
// W staged in LDS K-contiguous so B fragments load as ds_load_b128.
// Grid-stride over 16-row tiles; B fragments built once per wave.
// ---------------------------------------------------------------------------
template <int KPAD, bool KHALF>
__global__ __launch_bounds__(256) void gemm_xlxr(
    const float* __restrict__ X, int ldx, int fin, int HC,
    const float* __restrict__ Wl, const float* __restrict__ bl,
    const float* __restrict__ Wr, const float* __restrict__ br,
    float* __restrict__ xl, float* __restrict__ xr, int nrows)
{
    // [mat][n][k]: K-contiguous rows, each row KPAD*2 bytes (>=64B aligned)
    __shared__ _Float16 sW[2][16][KPAD];
    __shared__ float    sb[2][16];

    int tid = threadIdx.x;
    for (int i = tid; i < KPAD * 16; i += 256) {
        int n = i >> (KPAD == 128 ? 7 : 5);       // i / KPAD
        int k = i & (KPAD - 1);                   // i % KPAD
        sW[0][n][k] = (_Float16)((k < fin && n < HC) ? Wl[k * HC + n] : 0.0f);
        sW[1][n][k] = (_Float16)((k < fin && n < HC) ? Wr[k * HC + n] : 0.0f);
    }
    if (tid < 16) {
        sb[0][tid] = (tid < HC) ? bl[tid] : 0.0f;
        sb[1][tid] = (tid < HC) ? br[tid] : 0.0f;
    }
    __syncthreads();

    int wave  = tid >> 5;
    int lane  = tid & 31;
    int lidx  = lane & 15;            // A: row within tile, B/D: column
    int kbase = (lane >> 4) * 8;      // K sub-block per lane half
    int rhalf = (lane >> 4) * 8;      // D: row offset per lane half

    constexpr int NFRAG = KPAD / 32;
    v16h Bl[NFRAG], Br[NFRAG];
#pragma unroll
    for (int f = 0; f < NFRAG; ++f) {
        int koff = f * 32;
        v8h l0 = *(const v8h*)&sW[0][lidx][koff + kbase];
        v8h l1 = *(const v8h*)&sW[0][lidx][koff + 16 + kbase];
        v8h r0 = *(const v8h*)&sW[1][lidx][koff + kbase];
        v8h r1 = *(const v8h*)&sW[1][lidx][koff + 16 + kbase];
        Bl[f] = __builtin_shufflevector(l0, l1, 0,1,2,3,4,5,6,7,8,9,10,11,12,13,14,15);
        Br[f] = __builtin_shufflevector(r0, r1, 0,1,2,3,4,5,6,7,8,9,10,11,12,13,14,15);
    }
    float biasl = sb[0][lidx], biasr = sb[1][lidx];

    int ntiles = nrows >> 4;
    for (int tile = blockIdx.x * 8 + wave; tile < ntiles; tile += gridDim.x * 8) {
        int row0 = tile * 16;
        v8f accl = {}; v8f accr = {};
        const float* xrow = X + (size_t)(row0 + lidx) * ldx;
#pragma unroll
        for (int f = 0; f < NFRAG; ++f) {
            int koff = f * 32;
            // unguarded, 16B-aligned wide loads (rows are zero-padded)
            float4 r0 = *(const float4*)(xrow + koff + kbase);
            float4 r1 = *(const float4*)(xrow + koff + kbase + 4);
            v16h a;
            a[0] = (_Float16)r0.x; a[1] = (_Float16)r0.y;
            a[2] = (_Float16)r0.z; a[3] = (_Float16)r0.w;
            a[4] = (_Float16)r1.x; a[5] = (_Float16)r1.y;
            a[6] = (_Float16)r1.z; a[7] = (_Float16)r1.w;
            if (!KHALF) {
                float4 r2 = *(const float4*)(xrow + koff + 16 + kbase);
                float4 r3 = *(const float4*)(xrow + koff + 16 + kbase + 4);
                a[8]  = (_Float16)r2.x; a[9]  = (_Float16)r2.y;
                a[10] = (_Float16)r2.z; a[11] = (_Float16)r2.w;
                a[12] = (_Float16)r3.x; a[13] = (_Float16)r3.y;
                a[14] = (_Float16)r3.z; a[15] = (_Float16)r3.w;
            } else {
#pragma unroll
                for (int i = 8; i < 16; ++i) a[i] = (_Float16)0.0f;
            }
            accl = __builtin_amdgcn_wmma_f32_16x16x32_f16(false, a, false, Bl[f],
                                                          (short)0, accl, false, false);
            accr = __builtin_amdgcn_wmma_f32_16x16x32_f16(false, a, false, Br[f],
                                                          (short)0, accr, false, false);
        }
        // D layout: lane -> column (lane&15), VGPR v -> row v + 8*(lane>>4)
#pragma unroll
        for (int v = 0; v < 8; ++v) {
            size_t r = (size_t)(row0 + rhalf + v);
            xl[r * 16 + lidx] = accl[v] + biasl;
            xr[r * 16 + lidx] = accr[v] + biasr;
        }
    }
}

// ---------------------------------------------------------------------------
// Monotonic float<->uint encoding for atomicMax-based segment_max
// ---------------------------------------------------------------------------
__device__ __forceinline__ unsigned enc_f32(float f) {
    unsigned u = __float_as_uint(f);
    return (u & 0x80000000u) ? ~u : (u | 0x80000000u);
}

__global__ void decode_max(unsigned* __restrict__ p, size_t n) {
    size_t i = (size_t)blockIdx.x * blockDim.x + threadIdx.x;
    if (i >= n) return;
    unsigned u = p[i];
    float f = 0.0f;
    if (u != 0u) {
        unsigned bits = (u & 0x80000000u) ? (u ^ 0x80000000u) : ~u;
        if ((bits & 0x7F800000u) != 0x7F800000u) f = __uint_as_float(bits);
    }
    ((float*)p)[i] = f;   // in-place: encoded uint -> float (non-finite -> 0)
}

// ---------------------------------------------------------------------------
// Load a padded-16 node row (64B aligned) with wide loads into registers
// ---------------------------------------------------------------------------
template <int NQ>
__device__ __forceinline__ void load_row(const float* __restrict__ base, float* la) {
    const float4* p4 = (const float4*)base;
#pragma unroll
    for (int q = 0; q < NQ; ++q) {
        float4 v = p4[q];
        la[4 * q + 0] = v.x; la[4 * q + 1] = v.y;
        la[4 * q + 2] = v.z; la[4 * q + 3] = v.w;
    }
}

// ---------------------------------------------------------------------------
// Edge pass A: alpha[e,h] = sum_c leakyrelu(xl[src]+xr[dst]+ef, 0.2)*att[h,c]
//              amax[dst,h] = atomic max (encoded)
// ---------------------------------------------------------------------------
template <int C>
__global__ __launch_bounds__(256) void edge_alpha(
    const int* __restrict__ src, const int* __restrict__ dst,
    const float* __restrict__ ea,
    const float* __restrict__ xl, const float* __restrict__ xr,
    const float* __restrict__ We, const float* __restrict__ att,
    float* __restrict__ alpha, unsigned* __restrict__ amax, int E)
{
    constexpr int HC = 3 * C;
    constexpr int NQ = (HC + 3) / 4;
    __shared__ float sWe[3 * HC], sAtt[HC];
    if (threadIdx.x < 3 * HC) sWe[threadIdx.x] = We[threadIdx.x];
    if (threadIdx.x < HC)     sAtt[threadIdx.x] = att[threadIdx.x];
    __syncthreads();

    int e = blockIdx.x * blockDim.x + threadIdx.x;
    if (e >= E) return;
    int s = src[e], d = dst[e];
    float e0 = ea[(size_t)e * 3 + 0];
    float e1 = ea[(size_t)e * 3 + 1];
    float e2 = ea[(size_t)e * 3 + 2];
    float la[NQ * 4], ra[NQ * 4];
    load_row<NQ>(xl + (size_t)s * 16, la);
    load_row<NQ>(xr + (size_t)d * 16, ra);
#pragma unroll
    for (int h = 0; h < 3; ++h) {
        float acc = 0.0f;
#pragma unroll
        for (int c = 0; c < C; ++c) {
            int j = h * C + c;
            float ef = e0 * sWe[j] + e1 * sWe[HC + j] + e2 * sWe[2 * HC + j];
            float v = la[j] + ra[j] + ef;
            v = v > 0.0f ? v : NEG_SLOPE * v;
            acc += v * sAtt[j];
        }
        alpha[(size_t)e * 3 + h] = acc;
        atomicMax(&amax[(size_t)d * 4 + h], enc_f32(acc));
    }
}

// ---------------------------------------------------------------------------
// Edge pass B (fused softmax-aggregate): ex = exp(alpha - amax[dst]);
// den[dst,h] += ex;  hout[dst, h*C+c] += ex * xl[src, h*C+c]
// (normalization a = ex/(den+eps) folded into node_finish)
// ---------------------------------------------------------------------------
template <int C>
__global__ __launch_bounds__(256) void edge_accum(
    const int* __restrict__ src, const int* __restrict__ dst,
    const float* __restrict__ xl, const float* __restrict__ alpha,
    const float* __restrict__ amaxf, float* __restrict__ den,
    float* __restrict__ hout, int E)
{
    constexpr int HC = 3 * C;
    constexpr int NQ = (HC + 3) / 4;
    int e = blockIdx.x * blockDim.x + threadIdx.x;
    if (e >= E) return;
    int s = src[e], d = dst[e];
    float la[NQ * 4];
    load_row<NQ>(xl + (size_t)s * 16, la);
    float* po = hout + (size_t)d * 16;
#pragma unroll
    for (int h = 0; h < 3; ++h) {
        float ex = __expf(alpha[(size_t)e * 3 + h] - amaxf[(size_t)d * 4 + h]);
        atomicAdd(&den[(size_t)d * 4 + h], ex);
#pragma unroll
        for (int c = 0; c < C; ++c) {
            int j = h * C + c;
            atomicAdd(&po[j], ex * la[j]);
        }
    }
}

// ---------------------------------------------------------------------------
// Node finish: h = hout/(den+1e-16) + bias ; optional leaky(0.01); zero-pad
// ---------------------------------------------------------------------------
template <int C>
__global__ __launch_bounds__(256) void node_finish(
    float* __restrict__ hout, const float* __restrict__ den,
    const float* __restrict__ bias, int N, int act)
{
    constexpr int HC = 3 * C;
    int n = blockIdx.x * blockDim.x + threadIdx.x;
    if (n >= N) return;
    float* p = hout + (size_t)n * 16;
    float la[16];
    load_row<4>(p, la);
#pragma unroll
    for (int j = 0; j < 16; ++j) {
        float v = 0.0f;
        if (j < HC) {
            v = la[j] / (den[(size_t)n * 4 + (j / C)] + 1e-16f) + bias[j];
            if (act) v = v > 0.0f ? v : OUT_SLOPE * v;
        }
        la[j] = v;
    }
    float4* p4 = (float4*)p;
#pragma unroll
    for (int q = 0; q < 4; ++q) {
        float4 v; v.x = la[4 * q]; v.y = la[4 * q + 1];
        v.z = la[4 * q + 2]; v.w = la[4 * q + 3];
        p4[q] = v;
    }
}

// ---------------------------------------------------------------------------
// Gate pooling
// ---------------------------------------------------------------------------
__global__ __launch_bounds__(256) void gate_kernel(
    const float* __restrict__ h, const int* __restrict__ batch,
    const float* __restrict__ gW, const float* __restrict__ gb,
    float* __restrict__ gate, unsigned* __restrict__ gmax, int N)
{
    int n = blockIdx.x * blockDim.x + threadIdx.x;
    if (n >= N) return;
    float la[16];
    load_row<4>(h + (size_t)n * 16, la);
    float g = gb[0];
#pragma unroll
    for (int j = 0; j < 15; ++j) g += la[j] * gW[j];
    gate[n] = g;
    atomicMax(&gmax[batch[n]], enc_f32(g));
}

__global__ __launch_bounds__(256) void gate_accum(
    const float* __restrict__ h, const int* __restrict__ batch,
    const float* __restrict__ gate, const float* __restrict__ gmaxf,
    float* __restrict__ gden, float* __restrict__ gsum, int N)
{
    int n = blockIdx.x * blockDim.x + threadIdx.x;
    if (n >= N) return;
    int b = batch[n];
    float ge = __expf(gate[n] - gmaxf[b]);
    atomicAdd(&gden[b], ge);
    float la[16];
    load_row<4>(h + (size_t)n * 16, la);
#pragma unroll
    for (int j = 0; j < 15; ++j) atomicAdd(&gsum[(size_t)b * 16 + j], ge * la[j]);
}

__global__ void final_mlp(const float* __restrict__ gsum, const float* __restrict__ gden,
                          const float* __restrict__ fW1, const float* __restrict__ fb1,
                          const float* __restrict__ fW2, const float* __restrict__ fb2,
                          float* __restrict__ out, int NB)
{
    int b = threadIdx.x;
    if (b >= NB) return;
    float g[15];
#pragma unroll
    for (int j = 0; j < 15; ++j) g[j] = gsum[(size_t)b * 16 + j] / (gden[b] + 1e-16f);
    float o = fb2[0];
#pragma unroll
    for (int k = 0; k < 10; ++k) {
        float t = fb1[k];
#pragma unroll
        for (int j = 0; j < 15; ++j) t += g[j] * fW1[j * 10 + k];
        t = t > 0.0f ? t : OUT_SLOPE * t;
        o += t * fW2[k];
    }
    out[b] = o;
}

// ---------------------------------------------------------------------------
// Host-side per-layer driver
// ---------------------------------------------------------------------------
template <int C, int KPAD, bool KHALF>
static void run_layer(const float* hin, int ldx, int fin, float* hout_buf,
                      const float* Wl, const float* bl, const float* Wr, const float* br,
                      const float* We, const float* att, const float* bias,
                      float* xl, float* xr, unsigned* amax, float* den, float* alpha,
                      const int* src, const int* dst, const float* ea,
                      int N, int E, int act, hipStream_t stream)
{
    gemm_xlxr<KPAD, KHALF><<<1024, 256, 0, stream>>>(hin, ldx, fin, 3 * C,
                                                     Wl, bl, Wr, br, xl, xr, N);
    // amax (4N u32) and den (4N f32) are contiguous in ws
    fill_zero_u32<<<2048, 256, 0, stream>>>(amax, (size_t)8 * N);
    fill_zero_u32<<<2048, 256, 0, stream>>>((unsigned*)hout_buf, (size_t)16 * N);
    edge_alpha<C><<<(E + 255) / 256, 256, 0, stream>>>(src, dst, ea, xl, xr, We, att,
                                                       alpha, amax, E);
    decode_max<<<(4 * N + 255) / 256, 256, 0, stream>>>(amax, (size_t)4 * N);
    edge_accum<C><<<(E + 255) / 256, 256, 0, stream>>>(src, dst, xl, alpha,
                                                       (const float*)amax, den, hout_buf, E);
    node_finish<C><<<(N + 255) / 256, 256, 0, stream>>>(hout_buf, den, bias, N, act);
}

extern "C" void kernel_launch(void* const* d_in, const int* in_sizes, int n_in,
                              void* d_out, int out_size, void* d_ws, size_t ws_size,
                              hipStream_t stream)
{
    const float* x     = (const float*)d_in[0];
    const int*   ei    = (const int*)d_in[1];
    const float* ea    = (const float*)d_in[2];
    const int*   batch = (const int*)d_in[3];

    const int F = 128;
    const int N = in_sizes[0] / F;       // 262144
    const int E = in_sizes[1] / 2;       // 4194304
    const int NB = out_size;             // 64

    const float *Wl[4], *bl[4], *Wr[4], *br[4], *We[4], *att[4], *cb[4];
    for (int l = 0; l < 4; ++l) {
        int base = 4 + 7 * l;
        Wl[l]  = (const float*)d_in[base + 0];
        bl[l]  = (const float*)d_in[base + 1];
        Wr[l]  = (const float*)d_in[base + 2];
        br[l]  = (const float*)d_in[base + 3];
        We[l]  = (const float*)d_in[base + 4];
        att[l] = (const float*)d_in[base + 5];
        cb[l]  = (const float*)d_in[base + 6];
    }
    const float* gW  = (const float*)d_in[32];
    const float* gb  = (const float*)d_in[33];
    const float* fW1 = (const float*)d_in[34];
    const float* fb1 = (const float*)d_in[35];
    const float* fW2 = (const float*)d_in[36];
    const float* fb2 = (const float*)d_in[37];

    // Workspace layout (32-bit words)
    float* ws = (float*)d_ws;
    size_t o = 0;
    float*    hA    = ws + o; o += (size_t)16 * N;
    float*    hB    = ws + o; o += (size_t)16 * N;
    float*    xl    = ws + o; o += (size_t)16 * N;
    float*    xr    = ws + o; o += (size_t)16 * N;
    unsigned* amax  = (unsigned*)(ws + o); o += (size_t)4 * N;   // den follows
    float*    den   = ws + o; o += (size_t)4 * N;
    float*    alpha = ws + o; o += (size_t)3 * E;
    float*    gate  = ws + o; o += (size_t)N;
    unsigned* gmax  = (unsigned*)(ws + o); o += 64;              // gden, gsum follow
    float*    gden  = ws + o; o += 64;
    float*    gsum  = ws + o; o += 1024;

    const int* src = ei;
    const int* dst = ei + E;

    // Layer 1: fin=128, C=3 -> hA
    run_layer<3, 128, false>(x,  128, 128, hA, Wl[0], bl[0], Wr[0], br[0], We[0], att[0], cb[0],
                             xl, xr, amax, den, alpha, src, dst, ea, N, E, /*act=*/0, stream);
    // Layer 2: fin=9, C=5 -> hB, leaky(0.01) after
    run_layer<5, 32, true>(hA, 16, 9,  hB, Wl[1], bl[1], Wr[1], br[1], We[1], att[1], cb[1],
                           xl, xr, amax, den, alpha, src, dst, ea, N, E, /*act=*/1, stream);
    // Layer 3: fin=15, C=5 -> hA
    run_layer<5, 32, true>(hB, 16, 15, hA, Wl[2], bl[2], Wr[2], br[2], We[2], att[2], cb[2],
                           xl, xr, amax, den, alpha, src, dst, ea, N, E, /*act=*/0, stream);
    // Layer 4: fin=15, C=5 -> hB, leaky(0.01) after
    run_layer<5, 32, true>(hA, 16, 15, hB, Wl[3], bl[3], Wr[3], br[3], We[3], att[3], cb[3],
                           xl, xr, amax, den, alpha, src, dst, ea, N, E, /*act=*/1, stream);

    // Gate pooling over batch (B = NB segments)
    fill_zero_u32<<<1, 256, 0, stream>>>(gmax, 64 + 64 + 1024);
    gate_kernel<<<N / 256, 256, 0, stream>>>(hB, batch, gW, gb, gate, gmax, N);
    decode_max<<<1, 64, 0, stream>>>(gmax, 64);
    gate_accum<<<N / 256, 256, 0, stream>>>(hB, batch, gate, (const float*)gmax,
                                            gden, gsum, N);
    final_mlp<<<1, 64, 0, stream>>>(gsum, gden, fW1, fb1, fW2, fb2, (float*)d_out, NB);
}
